// LocalToPair_65335042506785
// MI455X (gfx1250) — compile-verified, matrix operand order checked by reading, not run on hardware
//
#include <hip/hip_runtime.h>
#include <hip/hip_bf16.h>

typedef __attribute__((ext_vector_type(16))) __bf16 v16bf;
typedef __attribute__((ext_vector_type(8)))  __bf16 v8bf;
typedef __attribute__((ext_vector_type(8)))  float  v8f;

#define N_    512
#define L_    256
#define P_    64
#define TP    128      // 2*P
#define KOUT  192      // 3*P
#define LN_EPS 1e-5f

// jax.nn.gelu (approximate=True), branch-free:
// 0.5*x*(1+tanh(u)) == x * sigmoid(2u),  2u = 2*sqrt(2/pi)*(x + 0.044715 x^3)
// sigmoid via v_exp_f32 + v_rcp_f32 (TRANS ops co-execute with XDL WMMA).
__device__ __forceinline__ float gelu_fast(float x) {
    float z = 1.5957691216057308f * x * (1.0f + 0.044715f * x * x);  // 2u
    float e = __builtin_amdgcn_exp2f(-1.4426950408889634f * z);      // exp(-2u)
    return x * __builtin_amdgcn_rcpf(1.0f + e);
}

// Load a 16x32 bf16 WMMA fragment (A) from row-major LDS [row][stride],
// or a 32x16 B fragment from K-contiguous (transposed) LDS [n][stride].
// Per ISA 7.12.2: lane<16 holds K=k0..k0+7 (v0..3) and K=k0+16..23 (v4..7);
// lane>=16 holds K=k0+8..15 and K=k0+24..31.
__device__ __forceinline__ v16bf load_frag(const __bf16* base, int stride,
                                           int row0, int k0, int lane) {
    int r  = lane & 15;
    int hi = lane >> 4;
    const __bf16* p = base + (size_t)(row0 + r) * stride + k0 + 8 * hi;
    v16bf f;
    ((v8bf*)&f)[0] = *(const v8bf*)(p);
    ((v8bf*)&f)[1] = *(const v8bf*)(p + 16);
    return f;
}

// ---------------------------------------------------------------------------
// Kernel 1: l = LN(local); lg/lv/rg/rv = l @ W_{left,right}_{gate,value}
// grid = 512 (one local row per block), block = 256
// ---------------------------------------------------------------------------
__global__ void __launch_bounds__(256) prep_kernel(
    const float* __restrict__ local,
    const float* __restrict__ Wlg, const float* __restrict__ Wlv,
    const float* __restrict__ Wrg, const float* __restrict__ Wrv,
    float* __restrict__ lg, float* __restrict__ lv,
    float* __restrict__ rg, float* __restrict__ rv) {

    __shared__ float lrow[L_];
    __shared__ float red[256];

    const int i = blockIdx.x;
    const int t = threadIdx.x;

    float x = local[(size_t)i * L_ + t];
    red[t] = x;
    __syncthreads();
    for (int s = 128; s > 0; s >>= 1) { if (t < s) red[t] += red[t + s]; __syncthreads(); }
    float mu = red[0] * (1.0f / L_);
    __syncthreads();
    float d = x - mu;
    red[t] = d * d;
    __syncthreads();
    for (int s = 128; s > 0; s >>= 1) { if (t < s) red[t] += red[t + s]; __syncthreads(); }
    float rs = rsqrtf(red[0] * (1.0f / L_) + LN_EPS);
    lrow[t] = d * rs;
    __syncthreads();

    const int c = t & 127;
    const float* Wg = (t < 128) ? Wlg : Wrg;
    const float* Wv = (t < 128) ? Wlv : Wrv;
    float ag = 0.f, av = 0.f;
    for (int k = 0; k < L_; ++k) {
        float v = lrow[k];
        ag += v * Wg[k * TP + c];
        av += v * Wv[k * TP + c];
    }
    if (t < 128) { lg[(size_t)i * TP + c] = ag; lv[(size_t)i * TP + c] = av; }
    else         { rg[(size_t)i * TP + c] = ag; rv[(size_t)i * TP + c] = av; }
}

// ---------------------------------------------------------------------------
// Kernel 2: per 16x16 pair tile: p = LN(pair); gate/value GEMMs via WMMA;
// gelu-gated masked products; accumulate left (sum over j) and right (sum
// over i) with f32 atomics.  grid = (32,32), block = 256 (8 waves).
// ---------------------------------------------------------------------------
__global__ void __launch_bounds__(256) pass1_kernel(
    const float* __restrict__ pair, const int* __restrict__ mask,
    const float* __restrict__ Wpg, const float* __restrict__ Wpv,
    const float* __restrict__ lg, const float* __restrict__ lv,
    const float* __restrict__ rg, const float* __restrict__ rv,
    float* __restrict__ left, float* __restrict__ right) {

    __shared__ __align__(16) __bf16 pbf[256 * P_];   // 32 KB, LN'd pair tile (bf16)
    __shared__ __align__(16) __bf16 WgT[TP * P_];    // 16 KB, W_pair_gate^T  [c][k]
    __shared__ __align__(16) __bf16 WvT[TP * P_];    // 16 KB, W_pair_value^T [c][k]
    __shared__ float lg_t[16][TP];                   // lg rows i0..i0+15
    __shared__ float rv_t[16][TP];                   // rv rows i0..i0+15
    __shared__ float lv_t[16][TP];                   // lv rows j0..j0+15
    __shared__ float rg_t[16][TP];                   // rg rows j0..j0+15
    __shared__ float left_acc[16][TP];
    __shared__ float mI[16], mJ[16];

    const int t  = threadIdx.x;
    const int i0 = blockIdx.y * 16;
    const int j0 = blockIdx.x * 16;

    // --- LN one pair element per thread: e = t = il*16 + jl
    {
        const int il = t >> 4, jl = t & 15;
        const float* src = pair + (((size_t)(i0 + il) * N_) + (j0 + jl)) * P_;
        float4 buf[16];
        float s = 0.f;
        for (int q = 0; q < 16; ++q) {
            buf[q] = ((const float4*)src)[q];
            s += buf[q].x + buf[q].y + buf[q].z + buf[q].w;
        }
        float mu = s * (1.0f / P_);
        float vs = 0.f;
        for (int q = 0; q < 16; ++q) {
            float a = buf[q].x - mu, b = buf[q].y - mu;
            float c = buf[q].z - mu, d = buf[q].w - mu;
            vs += a * a + b * b + c * c + d * d;
        }
        float rsd = rsqrtf(vs * (1.0f / P_) + LN_EPS);
        __bf16* dst = pbf + t * P_;
        for (int q = 0; q < 16; ++q) {
            dst[4 * q + 0] = (__bf16)((buf[q].x - mu) * rsd);
            dst[4 * q + 1] = (__bf16)((buf[q].y - mu) * rsd);
            dst[4 * q + 2] = (__bf16)((buf[q].z - mu) * rsd);
            dst[4 * q + 3] = (__bf16)((buf[q].w - mu) * rsd);
        }
    }
    // --- weights, transposed so B-fragments are K-contiguous
    for (int idx = t; idx < P_ * TP; idx += 256) {
        int k = idx >> 7, c = idx & 127;             // idx = k*128 + c
        WgT[c * P_ + k] = (__bf16)Wpg[idx];
        WvT[c * P_ + k] = (__bf16)Wpv[idx];
    }
    // --- broadcast rows + zero accumulators
    for (int idx = t; idx < 16 * TP; idx += 256) {
        int rr = idx >> 7, c = idx & 127;
        lg_t[rr][c] = lg[(size_t)(i0 + rr) * TP + c];
        rv_t[rr][c] = rv[(size_t)(i0 + rr) * TP + c];
        lv_t[rr][c] = lv[(size_t)(j0 + rr) * TP + c];
        rg_t[rr][c] = rg[(size_t)(j0 + rr) * TP + c];
        left_acc[rr][c] = 0.f;
    }
    if (t < 16) { mI[t] = (float)mask[i0 + t]; mJ[t] = (float)mask[j0 + t]; }
    __syncthreads();

    const int w = t >> 5, lane = t & 31;
    const int n0 = w * 16;                   // this wave owns columns n0..n0+15
    const int nl = lane & 15, hi = lane >> 4;
    const int col = n0 + nl;

    const v16bf bg0 = load_frag(WgT, P_, n0, 0, lane);
    const v16bf bg1 = load_frag(WgT, P_, n0, 32, lane);
    const v16bf bv0 = load_frag(WvT, P_, n0, 0, lane);
    const v16bf bv1 = load_frag(WvT, P_, n0, 32, lane);

    // j-indexed operands are loop-invariant per lane: hoist out of the M loop.
    float racc[8], pmj[8], lvv[8], rgv[8];
    for (int r = 0; r < 8; ++r) {
        int m = r + 8 * hi;                  // j_local of C element r
        racc[r] = 0.f;
        pmj[r]  = mJ[m];
        lvv[r]  = lv_t[m][col];
        rgv[r]  = rg_t[m][col];
    }

    for (int mt = 0; mt < 16; ++mt) {        // M-tile: i_local = mt, rows j_local
        v16bf a0 = load_frag(pbf, P_, mt * 16, 0, lane);
        v16bf a1 = load_frag(pbf, P_, mt * 16, 32, lane);
        v8f accg = {}, accv = {};
        accg = __builtin_amdgcn_wmma_f32_16x16x32_bf16(false, a0, false, bg0, (short)0, accg, false, false);
        accg = __builtin_amdgcn_wmma_f32_16x16x32_bf16(false, a1, false, bg1, (short)0, accg, false, false);
        accv = __builtin_amdgcn_wmma_f32_16x16x32_bf16(false, a0, false, bv0, (short)0, accv, false, false);
        accv = __builtin_amdgcn_wmma_f32_16x16x32_bf16(false, a1, false, bv1, (short)0, accv, false, false);

        const float mi  = mI[mt];
        const float lgv = lg_t[mt][col];
        const float rvv = rv_t[mt][col];
        float lsum = 0.f;
        for (int r = 0; r < 8; ++r) {
            float pm = mi * pmj[r];
            float pg = accg[r], pv = accv[r];
            float lG = gelu_fast(lgv + pg);          // gelu(lg[i] + pair_gate)
            float lV = lvv[r] + pv;                  // lv[j] + pair_value
            lsum += pm * lG * lV;
            float rG = gelu_fast(rgv[r] + pg);       // gelu(rg[j] + pair_gate)
            float rV = rvv + pv;                     // rv[i] + pair_value
            racc[r] += pm * rG * rV;
        }
        atomicAdd(&left_acc[mt][col], lsum);         // lanes l and l^16 share col
    }
    for (int r = 0; r < 8; ++r) {
        int m = r + 8 * hi;
        atomicAdd(&right[(size_t)(j0 + m) * TP + col], racc[r]);
    }
    __syncthreads();
    for (int idx = t; idx < 16 * TP; idx += 256) {
        int rr = idx >> 7, c = idx & 127;
        atomicAdd(&left[(size_t)(i0 + rr) * TP + c], left_acc[rr][c]);
    }
}

// ---------------------------------------------------------------------------
// Kernel 3: ppl = LN(left[i]+right[j]); out = [LN(pair), ppl] @ W_out
// grid = (32,32), block = 256 (8 waves).
// ---------------------------------------------------------------------------
__global__ void __launch_bounds__(256) pass2_kernel(
    const float* __restrict__ pair, const float* __restrict__ Wout,
    const float* __restrict__ left, const float* __restrict__ right,
    float* __restrict__ out) {

    __shared__ __align__(16) __bf16 abf[256 * KOUT]; // 96 KB: [p | ppl] bf16
    __shared__ __align__(16) __bf16 WoT[P_ * KOUT];  // 24 KB: W_out^T [c][k]
    __shared__ float left_t[16][TP];
    __shared__ float right_t[16][TP];

    const int t  = threadIdx.x;
    const int i0 = blockIdx.y * 16;
    const int j0 = blockIdx.x * 16;

    for (int idx = t; idx < 16 * TP; idx += 256) {
        int rr = idx >> 7, c = idx & 127;
        left_t[rr][c]  = left[(size_t)(i0 + rr) * TP + c];
        right_t[rr][c] = right[(size_t)(j0 + rr) * TP + c];
    }
    for (int idx = t; idx < KOUT * P_; idx += 256) {
        int k = idx >> 6, c = idx & 63;              // idx = k*64 + c
        WoT[c * KOUT + k] = (__bf16)Wout[idx];
    }
    __syncthreads();

    {
        const int il = t >> 4, jl = t & 15;
        // LN(pair) -> abf[:, 0:64]
        const float* src = pair + (((size_t)(i0 + il) * N_) + (j0 + jl)) * P_;
        float4 buf[16];
        float s = 0.f;
        for (int q = 0; q < 16; ++q) {
            buf[q] = ((const float4*)src)[q];
            s += buf[q].x + buf[q].y + buf[q].z + buf[q].w;
        }
        float mu = s * (1.0f / P_);
        float vs = 0.f;
        for (int q = 0; q < 16; ++q) {
            float a = buf[q].x - mu, b = buf[q].y - mu;
            float c = buf[q].z - mu, d = buf[q].w - mu;
            vs += a * a + b * b + c * c + d * d;
        }
        float rsd = rsqrtf(vs * (1.0f / P_) + LN_EPS);
        __bf16* dst = abf + t * KOUT;
        for (int q = 0; q < 16; ++q) {
            dst[4 * q + 0] = (__bf16)((buf[q].x - mu) * rsd);
            dst[4 * q + 1] = (__bf16)((buf[q].y - mu) * rsd);
            dst[4 * q + 2] = (__bf16)((buf[q].z - mu) * rsd);
            dst[4 * q + 3] = (__bf16)((buf[q].w - mu) * rsd);
        }
        // LN(left[i]+right[j]) -> abf[:, 64:192]  (two-pass over LDS)
        float s2 = 0.f, ss2 = 0.f;
        for (int c = 0; c < TP; ++c) {
            float x = left_t[il][c] + right_t[jl][c];
            s2 += x; ss2 += x * x;
        }
        float mu2  = s2 * (1.0f / TP);
        float var2 = ss2 * (1.0f / TP) - mu2 * mu2;
        float rs2  = rsqrtf(var2 + LN_EPS);
        __bf16* dst2 = dst + P_;
        for (int c = 0; c < TP; ++c) {
            float x = left_t[il][c] + right_t[jl][c];
            dst2[c] = (__bf16)((x - mu2) * rs2);
        }
    }
    __syncthreads();

    const int w = t >> 5, lane = t & 31;
    const int nl = lane & 15, hi = lane >> 4;

    for (int tile = w; tile < 64; tile += 8) {       // 16 M-tiles x 4 N-tiles
        int mt = tile >> 2, nt = tile & 3;
        int n0 = nt * 16;
        v8f acc = {};
        for (int kk = 0; kk < KOUT; kk += 32) {
            v16bf a = load_frag(abf, KOUT, mt * 16, kk, lane);
            v16bf b = load_frag(WoT, KOUT, n0, kk, lane);
            acc = __builtin_amdgcn_wmma_f32_16x16x32_bf16(false, a, false, b, (short)0, acc, false, false);
        }
        for (int r = 0; r < 8; ++r) {
            int m = r + 8 * hi;                      // e = mt*16+m -> il=mt, jl=m
            out[(((size_t)(i0 + mt)) * N_ + (j0 + m)) * P_ + n0 + nl] = acc[r];
        }
    }
}

// ---------------------------------------------------------------------------
extern "C" void kernel_launch(void* const* d_in, const int* in_sizes, int n_in,
                              void* d_out, int out_size, void* d_ws, size_t ws_size,
                              hipStream_t stream) {
    const float* local = (const float*)d_in[0];
    const float* pair  = (const float*)d_in[1];
    const int*   mask  = (const int*)d_in[2];
    const float* Wpg   = (const float*)d_in[3];
    const float* Wpv   = (const float*)d_in[4];
    const float* Wlg   = (const float*)d_in[5];
    const float* Wlv   = (const float*)d_in[6];
    const float* Wrg   = (const float*)d_in[7];
    const float* Wrv   = (const float*)d_in[8];
    const float* Wout  = (const float*)d_in[9];
    float* out = (float*)d_out;

    float* ws   = (float*)d_ws;
    const size_t M = (size_t)N_ * TP;   // 65536
    float* lg    = ws + 0 * M;
    float* lv    = ws + 1 * M;
    float* rg    = ws + 2 * M;
    float* rv    = ws + 3 * M;
    float* leftb = ws + 4 * M;
    float* rightb= ws + 5 * M;

    hipMemsetAsync(leftb, 0, 2 * M * sizeof(float), stream);  // left + right
    prep_kernel<<<N_, 256, 0, stream>>>(local, Wlg, Wlv, Wrg, Wrv, lg, lv, rg, rv);
    pass1_kernel<<<dim3(32, 32), 256, 0, stream>>>(pair, mask, Wpg, Wpv,
                                                   lg, lv, rg, rv, leftb, rightb);
    pass2_kernel<<<dim3(32, 32), 256, 0, stream>>>(pair, Wout, leftb, rightb, out);
}
